// AttentionEdgeWeighting_89386859364994
// MI455X (gfx1250) — matrix operand order repaired and checked by reading, not compiled, not run on hardware
//
#include <hip/hip_runtime.h>
#include <hip/hip_bf16.h>

// GAT edge attention, 5-kernel memory-bound pipeline for MI455X (gfx1250).
// Scores via v_wmma_f32_16x16x32_bf16 with the (tiny) weight as the A operand
// (block-diagonal rows selected branch-free via cndmask) and the streaming
// message/x_e tile as the B operand. Segment softmax via encoded-uint atomic
// max + f32 atomic add.

#define NHEADS 8
#define HDIM   32
#define MDIM   256   // NHEADS * HDIM

typedef __attribute__((ext_vector_type(16))) __bf16 v16bf;
typedef __attribute__((ext_vector_type(8)))  float  v8f;
typedef __attribute__((ext_vector_type(4)))  float  v4f;
typedef __attribute__((ext_vector_type(16))) float  v16f;

// ---- monotone float<->uint encoding so atomicMax(uint) == float max ----
__device__ __forceinline__ unsigned fenc(float f) {
    unsigned u = __float_as_uint(f);
    return u ^ (unsigned)(((int)u >> 31) | 0x80000000);
}
__device__ __forceinline__ float fdec(unsigned u) {
    unsigned m = ((int)u >= 0) ? 0xFFFFFFFFu : 0x80000000u;
    return __uint_as_float(u ^ m);
}

// Per-lane weight fragment: lane m (m = lane&15) caches the 16 bf16 weights it
// contributes to A-fragment hh == m (this code's k order, same as B: lane<16 K{0..15},
// lane>=16 K{16..31}). Rows m>=8 are never selected, so clamp the read.
__device__ __forceinline__ v16bf make_wfrag(const float* __restrict__ w,
                                            int woff, int lane) {
    const int m    = lane & 15;
    const int koff = (lane >> 4) * 16;
    const float* wp = w + (m & 7) * (2 * HDIM) + woff;
    v16bf f;
#pragma unroll
    for (int i = 0; i < 8; ++i) f[i]     = (__bf16)wp[koff + i];
#pragma unroll
    for (int i = 0; i < 8; ++i) f[8 + i] = (__bf16)wp[koff + 8 + i];
    return f;
}

// c[h] = sum_d X[r0 + (lane&15), h*32+d] * w[h, woff+d]  for h = 0..7,
// i.e. C'(16x16) = sum_hh A_hh(16x32 weights, row hh only) x B_hh(32x16 data).
// Lane n (n<16) ends up holding all 8 head scores of row r0+n in c[0..7].
__device__ __forceinline__ v8f score_tile(const float* __restrict__ X,
                                          const v16bf wfrag, int r0, int lane) {
    const int m  = lane & 15;            // data row / output column
    const int kb = (lane >> 4) * 16;     // B K sub-offset (assumed layout)
    const float* row = X + (size_t)(r0 + m) * MDIM + kb;
    // prefetch next tile's slice of this lane's stream
    __builtin_prefetch(row + (size_t)16 * MDIM, 0, 0);

    v8f c = {0.f, 0.f, 0.f, 0.f, 0.f, 0.f, 0.f, 0.f};
#pragma unroll
    for (int hh = 0; hh < NHEADS; ++hh) {
        // B fragment: 16 contiguous floats of this row's chunk -> bf16.
        const v16f f = *(const v16f*)(row + hh * HDIM);   // 4x global_load_b128
        v16bf b;
#pragma unroll
        for (int i = 0; i < 16; ++i) b[i] = (__bf16)f[i];

        // A fragment: weights, only row hh non-zero (branch-free select).
        v16bf a;
#pragma unroll
        for (int i = 0; i < 16; ++i) a[i] = (m == hh) ? wfrag[i] : (__bf16)0.0f;

        c = __builtin_amdgcn_wmma_f32_16x16x32_bf16(
                /*neg_a=*/false, a, /*neg_b=*/false, b,
                /*c_mod=*/(short)0, c, /*reuse_a=*/false, /*reuse_b=*/false);
    }
    return c;
}

// Pass 1: node_score[n][h] = x_e[n, h*32:h*32+32] . w[h, 32:64]
__global__ void k_node_score(const float* __restrict__ x_e,
                             const float* __restrict__ weight,
                             float* __restrict__ node_score, int N) {
    const int wave = blockIdx.x * (blockDim.x >> 5) + (threadIdx.x >> 5);
    const int r0 = wave * 16;
    if (r0 >= N) return;                       // uniform per wave
    const int lane = threadIdx.x & 31;
    const v16bf wfrag = make_wfrag(weight, HDIM, lane);
    v8f c = score_tile(x_e, wfrag, r0, lane);
    if (lane < 16) {
        float* dst = node_score + (size_t)(r0 + lane) * NHEADS;
#pragma unroll
        for (int h = 0; h < NHEADS; ++h) dst[h] = c[h];   // 2x b128 stores
    }
}

// Pass 2: alpha = leakyrelu(m.w1 + node_score[target]); stash alpha; atomic max.
__global__ void k_edge_score(const float* __restrict__ msg,
                             const int* __restrict__ target,
                             const float* __restrict__ weight,
                             const float* __restrict__ node_score,
                             float* __restrict__ wsAlpha,
                             unsigned* __restrict__ mxEnc, int E) {
    const int wave = blockIdx.x * (blockDim.x >> 5) + (threadIdx.x >> 5);
    const int r0 = wave * 16;
    if (r0 >= E) return;
    const int lane = threadIdx.x & 31;
    const v16bf wfrag = make_wfrag(weight, 0, lane);
    v8f c = score_tile(msg, wfrag, r0, lane);
    if (lane < 16) {
        const int e = r0 + lane;
        const int t = target[e];
        const float* ns = node_score + (size_t)t * NHEADS;  // 2x b128 loads
        float* wa = wsAlpha + (size_t)e * NHEADS;
        unsigned* mp = mxEnc + (size_t)t * NHEADS;
#pragma unroll
        for (int h = 0; h < NHEADS; ++h) {
            float sc = c[h] + ns[h];
            float al = sc >= 0.f ? sc : 0.1f * sc;          // LeakyReLU(0.1)
            wa[h] = al;
            atomicMax(&mp[h], fenc(al));
        }
    }
}

// Init the segment-max (encoded) and segment-sum arrays.
__global__ void k_init(unsigned* __restrict__ mxEnc, float* __restrict__ s, int n) {
    const int id = blockIdx.x * blockDim.x + threadIdx.x;
    if (id < n) { mxEnc[id] = 0u; s[id] = 0.f; }
}

// Pass 3: ex = exp(alpha - max); stash ex in place; atomic sum per (target,h).
__global__ void k_exp_sum(float* __restrict__ wsAlpha,
                          const int* __restrict__ target,
                          const unsigned* __restrict__ mxEnc,
                          float* __restrict__ s, int EH) {
    const int id = blockIdx.x * blockDim.x + threadIdx.x;
    if (id >= EH) return;
    const int e = id >> 3;
    const int h = id & 7;
    const int t = target[e];
    const float mx = fdec(mxEnc[(size_t)t * NHEADS + h]);
    const float ex = __expf(wsAlpha[id] - mx);
    wsAlpha[id] = ex;
    atomicAdd(&s[(size_t)t * NHEADS + h], ex);
}

// Pass 4: out = m * ex / max(sum, 1e-16). Streaming, non-temporal.
__global__ void k_out(const float* __restrict__ msg,
                      const int* __restrict__ target,
                      const float* __restrict__ wsEx,
                      const float* __restrict__ s,
                      float* __restrict__ out, int E) {
    const long id = (long)blockIdx.x * blockDim.x + threadIdx.x; // E*64 float4 lanes
    const int e = (int)(id >> 6);
    if (e >= E) return;
    const int q = (int)(id & 63);
    const int h = q >> 3;
    const int t = target[e];
    const float ex = wsEx[(size_t)e * NHEADS + h];
    const float sc = ex / fmaxf(s[(size_t)t * NHEADS + h], 1e-16f);
    const v4f mv = __builtin_nontemporal_load(((const v4f*)msg) + id);
    v4f o = mv * sc;
    __builtin_nontemporal_store(o, ((v4f*)out) + id);
}

extern "C" void kernel_launch(void* const* d_in, const int* in_sizes, int n_in,
                              void* d_out, int out_size, void* d_ws, size_t ws_size,
                              hipStream_t stream) {
    (void)n_in; (void)out_size; (void)ws_size;
    // inputs: source(unused), target, message, x_e, weight
    const int*   target = (const int*)d_in[1];
    const float* msg    = (const float*)d_in[2];
    const float* x_e    = (const float*)d_in[3];
    const float* weight = (const float*)d_in[4];
    const int E = in_sizes[1];
    const int N = in_sizes[3] / MDIM;
    const int NH = N * NHEADS;
    const int EH = E * NHEADS;

    // workspace layout
    char* ws = (char*)d_ws;
    float*    wsAlpha    = (float*)ws;                               // E*H floats
    unsigned* mxEnc      = (unsigned*)(ws + (size_t)EH * 4);         // N*H uints
    float*    sSum       = (float*)(ws + (size_t)EH * 4 + (size_t)NH * 4);
    float*    node_score = (float*)(ws + (size_t)EH * 4 + (size_t)NH * 8);
    float*    out        = (float*)d_out;

    const int TPB = 256;               // 8 wave32 waves
    const int wavesPerBlk = TPB / 32;

    // Pass 0: init max/sum arrays
    k_init<<<(NH + TPB - 1) / TPB, TPB, 0, stream>>>(mxEnc, sSum, NH);

    // Pass 1: per-node target-side scores (WMMA)
    {
        const int waves = (N + 15) / 16;
        const int blocks = (waves + wavesPerBlk - 1) / wavesPerBlk;
        k_node_score<<<blocks, TPB, 0, stream>>>(x_e, weight, node_score, N);
    }

    // Pass 2: per-edge scores + segment max (WMMA)
    {
        const int waves = (E + 15) / 16;
        const int blocks = (waves + wavesPerBlk - 1) / wavesPerBlk;
        k_edge_score<<<blocks, TPB, 0, stream>>>(msg, target, weight, node_score,
                                                 wsAlpha, mxEnc, E);
    }

    // Pass 3: exponentials + segment sum
    k_exp_sum<<<(EH + TPB - 1) / TPB, TPB, 0, stream>>>(wsAlpha, target, mxEnc, sSum, EH);

    // Pass 4: scale messages -> output
    {
        const long total = (long)E * 64;
        const long blocks = (total + TPB - 1) / TPB;
        k_out<<<(int)blocks, TPB, 0, stream>>>(msg, target, wsAlpha, sSum, out, E);
    }
}
